// MLPSampler_72086731096420
// MI455X (gfx1250) — matrix operand order repaired and checked by reading, not compiled, not run on hardware
//
#include <hip/hip_runtime.h>
#include <hip/hip_bf16.h>

// ---------------------------------------------------------------------------
// MLP autoregressive sampler scorer for MI455X (gfx1250, wave32, WMMA).
// Per step i: policy head (GEMM 256x512 + tiny head + log_softmax) and
// history MLP (31->512->512->256) fully fused per 64-row block, all GEMMs on
// v_wmma_f32_16x16x32_f16, intermediates kept in LDS (137KB of 320KB WGP LDS).
// ---------------------------------------------------------------------------

typedef _Float16 v16h __attribute__((ext_vector_type(16)));
typedef _Float16 v8h  __attribute__((ext_vector_type(8)));
typedef float    v8f  __attribute__((ext_vector_type(8)));

constexpr int NN = 65536;   // batch
constexpr int DD = 32;      // discrete dim (steps)
constexpr int EE = 256;     // embed dim
constexpr int HH = 512;     // hidden
constexpr int MT = 64;      // rows per workgroup
constexpr int LDA = 520;    // LDS row stride (halves) for 512-wide buffers (+8 pad)
constexpr int LDX = 40;     // LDS row stride for the 32-wide history input
constexpr int SMEM_BYTES = (MT * LDA * 2 + MT * 32) * (int)sizeof(_Float16);
// bufA[64][520] + bufB[64][520] + 64*32 halves for bufX

// ----- fragment loaders (contiguous k order, same for A and B) ----------------------
// A 16x32 f16: lane L<16 holds row L, K 0..15 contiguous; lane L>=16: K 16..31
__device__ __forceinline__ v16h load_a_frag(const _Float16* base, int lda,
                                            int kb, int lane) {
  const int r = lane & 15, hi = lane >> 4;
  const _Float16* q = base + r * lda + kb + hi * 16;
  v8h lo = *(const v8h*)(q);
  v8h hh = *(const v8h*)(q + 8);
  v16h f;
#pragma unroll
  for (int t = 0; t < 8; ++t) { f[t] = lo[t]; f[t + 8] = hh[t]; }
  return f;
}

// B 32x16 f16 (stored as W^T [Nout][Kin] row-major): lane L<16 holds col L,
// K 0..15 contiguous; lane L>=16 holds col L-16, K 16..31 contiguous.
__device__ __forceinline__ v16h load_b_frag(const _Float16* Bt, int ldb,
                                            int nb, int kb, int lane) {
  const int n = lane & 15, hi = lane >> 4;
  return *(const v16h*)(Bt + (size_t)(nb + n) * ldb + kb + hi * 16);
}

// One wave computes rows [m-tile of 16] x cols [n0,n1), 4 n-tiles at a time.
// A: row-major f16 (LDS or global), Bt: W^T [Nout][Kin] f16 global,
// C: row-major f16 out (LDS or global). bias f32, optional relu.
__device__ __forceinline__ void gemm_tiles(const _Float16* A, int lda,
                                           const _Float16* __restrict__ Bt, int ldb,
                                           const float* __restrict__ bias,
                                           _Float16* C, int ldc,
                                           int Kin, int n0, int n1, bool relu) {
  const int lane = threadIdx.x & 31;
  const int nloc = lane & 15;
  const int mtop = (lane >> 4) * 8;
  for (int nb = n0; nb < n1; nb += 64) {
    v8f acc[4] = {};
    for (int k = 0; k < Kin; k += 32) {
      v16h a = load_a_frag(A, lda, k, lane);
#pragma unroll
      for (int s = 0; s < 4; ++s) {
        v16h b = load_b_frag(Bt, ldb, nb + s * 16, k, lane);
        acc[s] = __builtin_amdgcn_wmma_f32_16x16x32_f16(
            false, a, false, b, (short)0, acc[s], false, false);
      }
    }
#pragma unroll
    for (int s = 0; s < 4; ++s) {
      const int n = nb + s * 16 + nloc;
      const float bv = bias[n];
#pragma unroll
      for (int v = 0; v < 8; ++v) {
        float x = acc[s][v] + bv;
        if (relu) x = fmaxf(x, 0.f);
        C[(size_t)(mtop + v) * ldc + n] = (_Float16)x;
      }
    }
  }
}

// ----- fused per-step kernel ------------------------------------------------
__global__ __launch_bounds__(256) void step_kernel(
    const _Float16* __restrict__ stateIn, _Float16* __restrict__ stateOut,
    const _Float16* __restrict__ oW1t, const float* __restrict__ ob1,
    const _Float16* __restrict__ oW2t, const float* __restrict__ ob2,
    const _Float16* __restrict__ mW1t, const float* __restrict__ mb1,
    const _Float16* __restrict__ mW2t, const float* __restrict__ mb2,
    const _Float16* __restrict__ mW3t, const float* __restrict__ mb3,
    const int* __restrict__ samples, float* __restrict__ out, int step) {
  extern __shared__ char smem[];
  _Float16* bufA = (_Float16*)smem;          // [MT][LDA]  (T, later h2)
  _Float16* bufB = bufA + MT * LDA;          // [MT][LDA]  (h1)
  _Float16* bufX = bufB + MT * LDA;          // [MT][LDX]  (masked history)

  const int tid = threadIdx.x;
  const int wave = tid >> 5;
  const int rowbase = blockIdx.x * MT;
  const int mt = wave & 3;       // m-tile (16 rows) within the 64-row block
  const int nh = wave >> 2;      // n half (0 = lower, 1 = upper)

  // Phase P1: T = relu(state @ oW1 + ob1)   [64,256]x[256,512] -> bufA
  gemm_tiles(stateIn + (size_t)(rowbase + mt * 16) * EE, EE, oW1t, EE, ob1,
             bufA + (size_t)mt * 16 * LDA, LDA, EE, nh * 256, nh * 256 + 256,
             true);
  __syncthreads();

  // Phase P2: logits = T @ oW2 + ob2; log_softmax (K=2), gather chosen.
  if (tid < MT) {
    float l0 = ob2[0], l1 = ob2[1];
    const _Float16* trow = bufA + (size_t)tid * LDA;
    for (int j = 0; j < HH; ++j) {
      const float t = (float)trow[j];
      l0 += t * (float)oW2t[j];
      l1 += t * (float)oW2t[HH + j];
    }
    const float mx = fmaxf(l0, l1);
    const float lse = mx + __logf(__expf(l0 - mx) + __expf(l1 - mx));
    const int row = rowbase + tid;
    const int c = samples[(size_t)row * DD + step];
    out[(size_t)row * DD + step] = (c ? l1 : l0) - lse;
  }

  if (step + 1 < DD) {
    // Build masked history input x[64][32] (cols <= step kept, col 31 = pad 0)
    for (int idx = tid; idx < MT * 32; idx += blockDim.x) {
      const int r = idx >> 5, c = idx & 31;
      float v = 0.f;
      if (c <= step && c < DD - 1)
        v = (float)samples[(size_t)(rowbase + r) * DD + c];
      bufX[r * LDX + c] = (_Float16)v;
    }
    __syncthreads();

    // H1: h1 = relu(x @ mW1[i] + mb1[i])   [64,32]x[32,512] -> bufB
    gemm_tiles(bufX + mt * 16 * LDX, LDX, mW1t + (size_t)step * HH * 32, 32,
               mb1 + (size_t)step * HH, bufB + (size_t)mt * 16 * LDA, LDA, 32,
               nh * 256, nh * 256 + 256, true);
    __syncthreads();

    // H2: h2 = relu(h1 @ mW2[i] + mb2[i])  [64,512]x[512,512] -> bufA
    gemm_tiles(bufB + (size_t)mt * 16 * LDA, LDA,
               mW2t + (size_t)step * HH * HH, HH, mb2 + (size_t)step * HH,
               bufA + (size_t)mt * 16 * LDA, LDA, HH, nh * 256, nh * 256 + 256,
               true);
    __syncthreads();

    // H3: state' = h2 @ mW3[i] + mb3[i]    [64,512]x[512,256] -> global f16
    gemm_tiles(bufA + (size_t)mt * 16 * LDA, LDA,
               mW3t + (size_t)step * EE * HH, HH, mb3 + (size_t)step * EE,
               stateOut + (size_t)(rowbase + mt * 16) * EE, EE, HH, nh * 128,
               nh * 128 + 128, false);
  }
}

// ----- prep kernels ---------------------------------------------------------
// W [Kin][Nout] f32 -> Wt [Nout][KinPad] f16 (zero-pad k >= Kin), batched.
__global__ void cvt_transpose(const float* __restrict__ W,
                              _Float16* __restrict__ Wt, int Kin, int Nout,
                              int KinPad, int bsi, int bso) {
  const int b = blockIdx.y;
  const float* Wb = W + (size_t)b * bsi;
  _Float16* Wtb = Wt + (size_t)b * bso;
  const int total = Nout * KinPad;
  for (int idx = blockIdx.x * blockDim.x + threadIdx.x; idx < total;
       idx += gridDim.x * blockDim.x) {
    const int n = idx / KinPad, k = idx % KinPad;
    const float v = (k < Kin) ? Wb[(size_t)k * Nout + n] : 0.f;
    Wtb[(size_t)n * KinPad + k] = (_Float16)v;
  }
}

__global__ void init_state(const float* __restrict__ init_h,
                           _Float16* __restrict__ st) {
  const size_t idx = (size_t)blockIdx.x * blockDim.x + threadIdx.x;
  st[idx] = (_Float16)init_h[idx & (EE - 1)];
}

// ----- host launch ----------------------------------------------------------
extern "C" void kernel_launch(void* const* d_in, const int* in_sizes, int n_in,
                              void* d_out, int out_size, void* d_ws,
                              size_t ws_size, hipStream_t stream) {
  const float* init_h = (const float*)d_in[0];
  const float* oW1 = (const float*)d_in[1];
  const float* ob1 = (const float*)d_in[2];
  const float* oW2 = (const float*)d_in[3];
  const float* ob2 = (const float*)d_in[4];
  const float* mW1 = (const float*)d_in[5];
  const float* mb1 = (const float*)d_in[6];
  const float* mW2 = (const float*)d_in[7];
  const float* mb2 = (const float*)d_in[8];
  const float* mW3 = (const float*)d_in[9];
  const float* mb3 = (const float*)d_in[10];
  const int* samples = (const int*)d_in[11];
  float* out = (float*)d_out;

  // Carve workspace (f16 state ping-pong + transposed f16 weights).
  char* ws = (char*)d_ws;
  auto carve = [&](size_t bytes) {
    char* p = ws;
    ws += (bytes + 255) & ~(size_t)255;
    return p;
  };
  _Float16* stateA = (_Float16*)carve((size_t)NN * EE * 2);
  _Float16* stateB = (_Float16*)carve((size_t)NN * EE * 2);
  _Float16* oW1t = (_Float16*)carve((size_t)HH * EE * 2);          // [512][256]
  _Float16* oW2t = (_Float16*)carve((size_t)2 * HH * 2);           // [2][512]
  _Float16* mW1t = (_Float16*)carve((size_t)(DD - 1) * HH * 32 * 2);   // [31][512][32]
  _Float16* mW2t = (_Float16*)carve((size_t)(DD - 1) * HH * HH * 2);   // [31][512][512]
  _Float16* mW3t = (_Float16*)carve((size_t)(DD - 1) * EE * HH * 2);   // [31][256][512]
  if ((size_t)(ws - (char*)d_ws) > ws_size) return;  // workspace too small

  (void)hipFuncSetAttribute((const void*)step_kernel,
                            hipFuncAttributeMaxDynamicSharedMemorySize,
                            SMEM_BYTES);

  // Weight conversion / transpose (all weights stay hot in 192MB L2 after).
  cvt_transpose<<<dim3(64, 1), 256, 0, stream>>>(oW1, oW1t, EE, HH, EE, 0, 0);
  cvt_transpose<<<dim3(8, 1), 256, 0, stream>>>(oW2, oW2t, HH, 2, HH, 0, 0);
  cvt_transpose<<<dim3(64, DD - 1), 256, 0, stream>>>(
      mW1, mW1t, DD - 1, HH, 32, (DD - 1) * HH, HH * 32);
  cvt_transpose<<<dim3(256, DD - 1), 256, 0, stream>>>(
      mW2, mW2t, HH, HH, HH, HH * HH, HH * HH);
  cvt_transpose<<<dim3(128, DD - 1), 256, 0, stream>>>(
      mW3, mW3t, HH, EE, HH, HH * EE, EE * HH);
  init_state<<<(NN * EE) / 256, 256, 0, stream>>>(init_h, stateA);

  // 32 sequential fused steps (state ping-pong).
  const int nblocks = NN / MT;  // 1024 workgroups x 256 threads (8 waves)
  _Float16* sIn = stateA;
  _Float16* sOut = stateB;
  for (int i = 0; i < DD; ++i) {
    step_kernel<<<nblocks, 256, SMEM_BYTES, stream>>>(
        sIn, sOut, oW1t, ob1, oW2t, ob2, mW1t, mb1, mW2t, mb2, mW3t, mb3,
        samples, out, i);
    _Float16* t = sIn; sIn = sOut; sOut = t;
  }
}
